// SBN_Steroids_63213328662908
// MI455X (gfx1250) — matrix operand order repaired and checked by reading, not compiled, not run on hardware
//
#include <hip/hip_runtime.h>
#include <hip/hip_bf16.h>
#include <stdint.h>

// ---------------------------------------------------------------------------
// Problem constants (match reference setup_inputs)
// ---------------------------------------------------------------------------
#define NX 2048
#define NZ 256
#define BB 512
#define SS 10
#define KK 25
#define TT 30
#define MROWS (SS * BB)        // 5120
#define NT_N  (NX / 16)        // 128 column tiles for big GEMMs
#define NT_M  (MROWS / 16)     // 320 row tiles

typedef _Float16 h8   __attribute__((ext_vector_type(8)));
typedef _Float16 v16h __attribute__((ext_vector_type(16)));
typedef float    v8f  __attribute__((ext_vector_type(8)));

// ---------------------------------------------------------------------------
// Deterministic hash RNG (splitmix64-style) -> uniform [0,1)
// ---------------------------------------------------------------------------
__device__ __forceinline__ float urand(unsigned a, unsigned b) {
  unsigned long long x = (((unsigned long long)a << 32) ^ b) + 0x9E3779B97F4A7C15ull;
  x ^= x >> 30; x *= 0xBF58476D1CE4E5B9ull;
  x ^= x >> 27; x *= 0x94D049BB133111EBull;
  x ^= x >> 31;
  return (float)(unsigned)((x >> 40) & 0xFFFFFFu) * (1.0f / 16777216.0f);
}

// ---------------------------------------------------------------------------
// WMMA fragment loaders (CDNA5 16-bit layouts, wave32)
// A 16x32: lanes 0-15 hold K=0..15; lanes 16-31 hold K=16..31 (contiguous 16).
//   caller passes p = rowBase + k0 + half*16; hi chunk at +8.
// B 32x16: lanes 0-15 hold K=0..15; lanes 16-31 hold K=16..31 (contiguous 16).
//   caller passes p = colBase + k0 + half*16.
// ---------------------------------------------------------------------------
__device__ __forceinline__ v16h ldA(const _Float16* p) {
  h8 lo = *(const h8*)(p);
  h8 hi = *(const h8*)(p + 8);
  return __builtin_shufflevector(lo, hi, 0,1,2,3,4,5,6,7,8,9,10,11,12,13,14,15);
}
__device__ __forceinline__ v16h ldB(const _Float16* p) {
  h8 lo = *(const h8*)(p);
  h8 hi = *(const h8*)(p + 8);
  return __builtin_shufflevector(lo, hi, 0,1,2,3,4,5,6,7,8,9,10,11,12,13,14,15);
}
#define WMMA_F16(a, b, c) \
  __builtin_amdgcn_wmma_f32_16x16x32_f16(false, (a), false, (b), (short)0, (c), false, false)

// ---------------------------------------------------------------------------
// f32 -> f16 conversion
// ---------------------------------------------------------------------------
__global__ void kCvt(const float* __restrict__ src, _Float16* __restrict__ dst, int n) {
  int i = blockIdx.x * 256 + threadIdx.x;
  if (i < n) dst[i] = (_Float16)src[i];
}

// ---------------------------------------------------------------------------
// init: pz = clip(sigmoid(z_bias)); log tables; zero the pelbo accumulator
// ---------------------------------------------------------------------------
__global__ void kInit(const float* __restrict__ z_bias,
                      float* __restrict__ lpz1, float* __restrict__ lpz0,
                      float* __restrict__ pelbo) {
  int z = threadIdx.x;
  if (z < NZ) {
    float a = fminf(fmaxf(z_bias[z], -30.f), 30.f);
    float p = 1.f / (1.f + __expf(-a));
    p = fminf(fmaxf(p, 1e-6f), 1.f - 1e-6f);
    lpz1[z] = __logf(p);
    lpz0[z] = log1pf(-p);
  }
  if (z == 0) *pelbo = 0.f;
}

// ---------------------------------------------------------------------------
// GEMM-Q: q = clip(sigmoid(xin @ U^T + z_bias)); sample z; write f16 copies.
// M=512, N=256, K=2048. Two 16x16 N-tiles per wave (A fragment reused).
// 256 wave-jobs -> 32 blocks of 8 waves.
// ---------------------------------------------------------------------------
__global__ void kGemmQ(const _Float16* __restrict__ Ah, const _Float16* __restrict__ Bh,
                       const float* __restrict__ z_bias,
                       float* __restrict__ q_out, float* __restrict__ samples_out,
                       float* __restrict__ Lq1, float* __restrict__ Lq0,
                       _Float16* __restrict__ szf, _Float16* __restrict__ szr) {
  int wave = threadIdx.x >> 5, lane = threadIdx.x & 31;
  int half = lane >> 4, r = lane & 15, kb = half * 8;
  int job = blockIdx.x * 8 + wave;         // 0..255
  int m0 = (job >> 3) * 16;                // batch row
  int n0 = (job & 7) * 32;                 // two z-column tiles
  const _Float16* arow  = Ah + (size_t)(m0 + r) * NX;
  const _Float16* brow0 = Bh + (size_t)(n0 + r) * NX;
  const _Float16* brow1 = Bh + (size_t)(n0 + 16 + r) * NX;
  v8f c0 = {}, c1 = {};
  for (int k0 = 0; k0 < NX; k0 += 32) {
    v16h a  = ldA(arow + k0 + half * 16);
    v16h b0 = ldB(brow0 + k0 + half * 16);
    v16h b1 = ldB(brow1 + k0 + half * 16);
    c0 = WMMA_F16(a, b0, c0);
    c1 = WMMA_F16(a, b1, c1);
  }
  for (int j = 0; j < 8; ++j) {
    int row = m0 + half * 8 + j;
    for (int tc = 0; tc < 2; ++tc) {
      int z = n0 + tc * 16 + r;
      float acc = tc ? c1[j] : c0[j];
      float aa = fminf(fmaxf(acc + z_bias[z], -30.f), 30.f);
      float p = 1.f / (1.f + __expf(-aa));
      p = fminf(fmaxf(p, 1e-6f), 1.f - 1e-6f);
      int qi = row * NZ + z;
      q_out[qi] = p;
      Lq1[qi] = __logf(p);
      Lq0[qi] = log1pf(-p);
      for (int s = 0; s < SS; ++s) {
        float u = urand(0x5EED0000u + (unsigned)s, (unsigned)qi);
        float smp = (u < p) ? 1.f : 0.f;
        int idx = (s * BB + row) * NZ + z;
        samples_out[idx] = smp;
        szf[idx] = (_Float16)smp;
        szr[idx] = (z < KK) ? (_Float16)0.f : (_Float16)smp;
      }
    }
  }
}

// ---------------------------------------------------------------------------
// Base log-prob sums over the (never-projected) columns z >= KK.
// ---------------------------------------------------------------------------
__global__ void kBase(const float* __restrict__ samples,
                      const float* __restrict__ Lq1, const float* __restrict__ Lq0,
                      const float* __restrict__ lpz1, const float* __restrict__ lpz0,
                      float* __restrict__ blq, float* __restrict__ blp) {
  int sb = blockIdx.x * 256 + threadIdx.x;
  if (sb >= MROWS) return;
  int b = sb & (BB - 1);
  float sq = 0.f, sp = 0.f;
  for (int z = KK; z < NZ; ++z) {
    bool on = samples[(size_t)sb * NZ + z] > 0.5f;
    sq += on ? Lq1[b * NZ + z] : Lq0[b * NZ + z];
    sp += on ? lpz1[z] : lpz0[z];
  }
  blq[sb] = sq; blp[sb] = sp;
}

// ---------------------------------------------------------------------------
// Big GEMM: acc = A(5120x256,f16) @ V(2048x256,f16)^T.
// Block = 8 waves covering one M-tile x 256 columns; the 8KB A tile is
// async-staged into LDS once (global_load_async_to_lds_b128, ASYNCcnt) and
// each wave computes two N-tiles, reusing the LDS A fragment for 2 WMMAs.
// MODE 0: store raw acc -> G_rest.   MODE 1: ps/xouts epilogue.
// ---------------------------------------------------------------------------
template <int MODE>
__global__ void kGemmBig(const _Float16* __restrict__ Ah, const _Float16* __restrict__ Vh,
                         const float* __restrict__ x_bias,
                         float* __restrict__ grest, float* __restrict__ ps,
                         float* __restrict__ xouts) {
  __shared__ __align__(16) _Float16 As[16 * NZ];   // 8 KB A tile shared by 8 waves
  int tid = threadIdx.x;
  int wave = tid >> 5, lane = tid & 31;
  int half = lane >> 4, r = lane & 15, kb = half * 8;
  int m0 = (blockIdx.x >> 3) * 16;                 // 320 M-tiles
  int n0 = (blockIdx.x & 7) * 256 + wave * 32;     // this wave's two 16-col tiles

  // async stage: A tile is one contiguous 8KB block (16 rows x 256 halves)
  {
    unsigned lbase = (unsigned)(uintptr_t)(void*)As + (unsigned)tid * 32u;
    unsigned long long ga =
        (unsigned long long)(uintptr_t)(Ah + (size_t)m0 * NZ) + (unsigned long long)tid * 32ull;
    asm volatile("global_load_async_to_lds_b128 %0, %1, off" :: "v"(lbase), "v"(ga) : "memory");
    asm volatile("global_load_async_to_lds_b128 %0, %1, off offset:16" :: "v"(lbase), "v"(ga) : "memory");
    asm volatile("s_wait_asynccnt 0" ::: "memory");
  }
  __syncthreads();

  const _Float16* arow  = As + (size_t)r * NZ;     // LDS reads (ds_load_b128)
  const _Float16* brow0 = Vh + (size_t)(n0 + r) * NZ;
  const _Float16* brow1 = Vh + (size_t)(n0 + 16 + r) * NZ;
  v8f c0 = {}, c1 = {};
  for (int k0 = 0; k0 < NZ; k0 += 32) {
    v16h a  = ldA(arow + k0 + half * 16);
    v16h b0 = ldB(brow0 + k0 + half * 16);
    v16h b1 = ldB(brow1 + k0 + half * 16);
    c0 = WMMA_F16(a, b0, c0);
    c1 = WMMA_F16(a, b1, c1);
  }
  for (int j = 0; j < 8; ++j) {
    int row = m0 + half * 8 + j;
    for (int tc = 0; tc < 2; ++tc) {
      int col = n0 + tc * 16 + r;
      float acc = tc ? c1[j] : c0[j];
      size_t idx = (size_t)row * NX + col;
      if (MODE == 0) {
        grest[idx] = acc;
      } else {
        float aa = fminf(fmaxf(acc + x_bias[col], -30.f), 30.f);
        float p = 1.f / (1.f + __expf(-aa));
        p = fminf(fmaxf(p, 1e-6f), 1.f - 1e-6f);
        ps[idx] = p;
        float u = urand(0xC0FFEEu, (unsigned)idx);
        xouts[idx] = (u < p) ? 1.f : 0.f;
      }
    }
  }
}

// ---------------------------------------------------------------------------
// GF(2) Gaussian elimination on [A|b] (25x257), generated from hash RNG.
// ---------------------------------------------------------------------------
__global__ void kRowReduce(float* __restrict__ Cmat, int t) {
  __shared__ float M[KK * 257];
  __shared__ int piv;
  __shared__ float flag[KK];
  int tid = threadIdx.x;
  for (int idx = tid; idx < KK * 257; idx += 256) {
    int r = idx / 257, c = idx % 257;
    float u = (c < 256) ? urand(0xA11CE000u + (unsigned)t, (unsigned)(r * 256 + c))
                        : urand(0xB0B00000u + (unsigned)t, (unsigned)r);
    M[idx] = (u < 0.5f) ? 1.0f : 0.0f;
  }
  __syncthreads();
  for (int i = 0; i < KK; ++i) {
    if (tid == 0) {
      int p = KK - 1;
      for (int r = KK - 1; r >= i; --r)
        if (M[r * 257 + i] > 0.5f) p = r;
      piv = p;
    }
    __syncthreads();
    int p = piv;
    if (p != i) {
      for (int c = tid; c < 257; c += 256) {
        float x = M[i * 257 + c], y = M[p * 257 + c];
        M[i * 257 + c] = y; M[p * 257 + c] = x;
      }
    }
    __syncthreads();
    if (tid < KK) flag[tid] = (tid != i && M[tid * 257 + i] > 0.5f) ? 1.f : 0.f;
    __syncthreads();
    for (int idx = tid; idx < KK * 257; idx += 256) {
      int r = idx / 257, c = idx % 257;
      if (flag[r] > 0.5f) {
        float s = M[idx] + M[i * 257 + c];
        M[idx] = (s > 1.5f) ? 0.0f : s;   // mod-2 on {0,1,2}
      }
    }
    __syncthreads();
  }
  for (int idx = tid; idx < KK * 257; idx += 256) Cmat[idx] = M[idx];
}

// ---------------------------------------------------------------------------
// proj[sb,kk] = mod( samples[sb,25:] . C[kk,25:] + b'[kk], 2 )   (parity)
// Padded to 32 columns (zeros) so the rank-update GEMM is one K=32 WMMA step.
// ---------------------------------------------------------------------------
__global__ void kProj(const float* __restrict__ samples, const float* __restrict__ Cmat,
                      _Float16* __restrict__ proj_h) {
  int gid = blockIdx.x * 256 + threadIdx.x;
  if (gid >= MROWS * 32) return;
  int sb = gid >> 5, kk = gid & 31;
  if (kk >= KK) { proj_h[gid] = (_Float16)0.f; return; }
  int par = (Cmat[kk * 257 + 256] > 0.5f) ? 1 : 0;
  const float* srow = samples + (size_t)sb * NZ;
  const float* crow = Cmat + kk * 257;
  for (int z = KK; z < NZ; ++z)
    par += ((srow[z] > 0.5f) && (crow[z] > 0.5f)) ? 1 : 0;
  proj_h[gid] = (_Float16)(float)(par & 1);
}

// ---------------------------------------------------------------------------
// Per-t log q / log p_z (base sums + projected first-25-column contribution).
// ---------------------------------------------------------------------------
__global__ void kPerT(const _Float16* __restrict__ proj_h,
                      const float* __restrict__ Lq1, const float* __restrict__ Lq0,
                      const float* __restrict__ lpz1, const float* __restrict__ lpz0,
                      const float* __restrict__ blq, const float* __restrict__ blp,
                      float* __restrict__ lqt, float* __restrict__ lpzt) {
  int sb = blockIdx.x * 256 + threadIdx.x;
  if (sb >= MROWS) return;
  int b = sb & (BB - 1);
  float sq = blq[sb], sp = blp[sb];
  for (int z = 0; z < KK; ++z) {
    bool on = (float)proj_h[sb * 32 + z] > 0.5f;
    sq += on ? Lq1[b * NZ + z] : Lq0[b * NZ + z];
    sp += on ? lpz1[z] : lpz0[z];
  }
  lqt[sb] = sq; lpzt[sb] = sp;
}

// ---------------------------------------------------------------------------
// Rank-25 update + fused Bernoulli log-lik epilogue.  One proj A fragment
// (16x32, zero-padded K) feeds two WMMAs (two N-tiles); row log-lik sums are
// shfl_xor-reduced into deterministic per-tile partials (no atomics).
// ---------------------------------------------------------------------------
__global__ void kRankUpdate(const _Float16* __restrict__ proj_h, const _Float16* __restrict__ Vh,
                            const float* __restrict__ x_bias, const float* __restrict__ xin,
                            const float* __restrict__ grest, float* __restrict__ partial) {
  int wave = threadIdx.x >> 5, lane = threadIdx.x & 31;
  int half = lane >> 4, r = lane & 15, kb = half * 8;
  int job = blockIdx.x * 8 + wave;          // 0..20479
  int m0 = (job >> 6) * 16;
  int npair = job & 63;
  int n0 = npair * 32;
  v8f c0, c1;
  for (int j = 0; j < 8; ++j) {
    size_t rb = (size_t)(m0 + half * 8 + j) * NX;
    c0[j] = grest[rb + n0 + r];
    c1[j] = grest[rb + n0 + 16 + r];
  }
  v16h a  = ldA(proj_h + (size_t)(m0 + r) * 32 + half * 16);
  v16h b0 = ldB(Vh + (size_t)(n0 + r) * NZ + half * 16);
  v16h b1 = ldB(Vh + (size_t)(n0 + 16 + r) * NZ + half * 16);
  c0 = WMMA_F16(a, b0, c0);
  c1 = WMMA_F16(a, b1, c1);
  for (int j = 0; j < 8; ++j) {
    int row = m0 + half * 8 + j;
    for (int tc = 0; tc < 2; ++tc) {
      int col = n0 + tc * 16 + r;
      float acc = tc ? c1[j] : c0[j];
      float aa = fminf(fmaxf(acc + x_bias[col], -13.8155f), 13.8155f);  // logit(1-1e-6)
      float xv = xin[(size_t)(row & (BB - 1)) * NX + col];
      float ll = -log1pf(__expf(-aa)) - (1.f - xv) * aa;
      ll += __shfl_xor(ll, 1, 16);
      ll += __shfl_xor(ll, 2, 16);
      ll += __shfl_xor(ll, 4, 16);
      ll += __shfl_xor(ll, 8, 16);
      if (r == 0) partial[(size_t)row * NT_N + npair * 2 + tc] = ll;
    }
  }
}

__global__ void kSumPartial(const float* __restrict__ partial, float* __restrict__ lpx) {
  int sb = blockIdx.x * 256 + threadIdx.x;
  if (sb >= MROWS) return;
  float s = 0.f;
  for (int j = 0; j < NT_N; ++j) s += partial[(size_t)sb * NT_N + j];
  lpx[sb] = s;
}

// ---------------------------------------------------------------------------
// term1 - term2 for one t:  -(Σ_b (Σ_s w·(-logp) - Σ_s w·(-logq)) / Σ_s w)
// ---------------------------------------------------------------------------
__global__ void kTerm(const float* __restrict__ lqt, const float* __restrict__ lpzt,
                      const float* __restrict__ lpx, float* __restrict__ pelbo) {
  __shared__ float red[BB];
  int b = threadIdx.x;
  float mq = -1e30f;
  for (int s = 0; s < SS; ++s) mq = fmaxf(mq, lqt[s * BB + b]);
  float sw = 0.f, dnum = 0.f;
  for (int s = 0; s < SS; ++s) {
    int sb = s * BB + b;
    float lq = lqt[sb];
    float lp = lpzt[sb] + lpx[sb];
    float w = __expf(lq - mq);
    sw += w;
    dnum += w * (lq - lp);        // (-lp) - (-lq)
  }
  red[b] = -dnum / sw;
  __syncthreads();
  for (int off = BB / 2; off > 0; off >>= 1) {
    if (b < off) red[b] += red[b + off];
    __syncthreads();
  }
  if (b == 0) *pelbo += red[0];
}

__global__ void kFinal(const float* __restrict__ pelbo, float* __restrict__ loss) {
  if (threadIdx.x == 0) loss[0] = -(*pelbo) / (float)TT;
}

// ---------------------------------------------------------------------------
// Workspace layout (bytes)
// ---------------------------------------------------------------------------
static constexpr size_t OFF_XIN_H = 0;
static constexpr size_t OFF_U_H   = OFF_XIN_H + (size_t)BB * NX * 2;
static constexpr size_t OFF_V_H   = OFF_U_H   + (size_t)NZ * NX * 2;
static constexpr size_t OFF_SZF   = OFF_V_H   + (size_t)NX * NZ * 2;
static constexpr size_t OFF_SZR   = OFF_SZF   + (size_t)MROWS * NZ * 2;
static constexpr size_t OFF_PROJ  = OFF_SZR   + (size_t)MROWS * NZ * 2;
static constexpr size_t OFF_LQ1   = OFF_PROJ  + (size_t)MROWS * 32 * 2;
static constexpr size_t OFF_LQ0   = OFF_LQ1   + (size_t)BB * NZ * 4;
static constexpr size_t OFF_LPZ1  = OFF_LQ0   + (size_t)BB * NZ * 4;
static constexpr size_t OFF_LPZ0  = OFF_LPZ1  + (size_t)NZ * 4;
static constexpr size_t OFF_BLQ   = OFF_LPZ0  + (size_t)NZ * 4;
static constexpr size_t OFF_BLP   = OFF_BLQ   + (size_t)MROWS * 4;
static constexpr size_t OFF_LQT   = OFF_BLP   + (size_t)MROWS * 4;
static constexpr size_t OFF_LPZT  = OFF_LQT   + (size_t)MROWS * 4;
static constexpr size_t OFF_LPX   = OFF_LPZT  + (size_t)MROWS * 4;
static constexpr size_t OFF_CMAT  = OFF_LPX   + (size_t)MROWS * 4;
static constexpr size_t OFF_PELBO = OFF_CMAT  + ((size_t)KK * 257 * 4 + 255) / 256 * 256;
static constexpr size_t OFF_PART  = OFF_PELBO + 256;
static constexpr size_t OFF_GREST = OFF_PART  + (size_t)MROWS * NT_N * 4;
// total = OFF_GREST + 5120*2048*4  ~= 53 MB

extern "C" void kernel_launch(void* const* d_in, const int* in_sizes, int n_in,
                              void* d_out, int out_size, void* d_ws, size_t ws_size,
                              hipStream_t stream) {
  (void)in_sizes; (void)n_in; (void)out_size; (void)ws_size;
  const float* xin    = (const float*)d_in[0];
  const float* U      = (const float*)d_in[1];
  const float* V      = (const float*)d_in[2];
  const float* x_bias = (const float*)d_in[3];
  const float* z_bias = (const float*)d_in[4];

  float* out     = (float*)d_out;
  float* q_out   = out;                                    // 512*256
  float* smp_out = q_out   + (size_t)BB * NZ;              // 10*512*256
  float* ps_out  = smp_out + (size_t)MROWS * NZ;           // 10*512*2048
  float* xo_out  = ps_out  + (size_t)MROWS * NX;
  float* loss    = xo_out  + (size_t)MROWS * NX;           // 1

  char* ws = (char*)d_ws;
  _Float16* xin_h  = (_Float16*)(ws + OFF_XIN_H);
  _Float16* U_h    = (_Float16*)(ws + OFF_U_H);
  _Float16* V_h    = (_Float16*)(ws + OFF_V_H);
  _Float16* szf    = (_Float16*)(ws + OFF_SZF);
  _Float16* szr    = (_Float16*)(ws + OFF_SZR);
  _Float16* proj_h = (_Float16*)(ws + OFF_PROJ);
  float* Lq1   = (float*)(ws + OFF_LQ1);
  float* Lq0   = (float*)(ws + OFF_LQ0);
  float* lpz1  = (float*)(ws + OFF_LPZ1);
  float* lpz0  = (float*)(ws + OFF_LPZ0);
  float* blq   = (float*)(ws + OFF_BLQ);
  float* blp   = (float*)(ws + OFF_BLP);
  float* lqt   = (float*)(ws + OFF_LQT);
  float* lpzt  = (float*)(ws + OFF_LPZT);
  float* lpx   = (float*)(ws + OFF_LPX);
  float* Cmat  = (float*)(ws + OFF_CMAT);
  float* pelbo = (float*)(ws + OFF_PELBO);
  float* part  = (float*)(ws + OFF_PART);
  float* grest = (float*)(ws + OFF_GREST);

  // f16 copies for WMMA operands
  kCvt<<<(BB * NX + 255) / 256, 256, 0, stream>>>(xin, xin_h, BB * NX);
  kCvt<<<(NZ * NX + 255) / 256, 256, 0, stream>>>(U, U_h, NZ * NX);
  kCvt<<<(NX * NZ + 255) / 256, 256, 0, stream>>>(V, V_h, NX * NZ);
  kInit<<<1, 256, 0, stream>>>(z_bias, lpz1, lpz0, pelbo);

  // q + sampling (WMMA, M=512 N=256 K=2048, 2 tiles/wave)
  kGemmQ<<<32, 256, 0, stream>>>(xin_h, U_h, z_bias, q_out, smp_out, Lq1, Lq0, szf, szr);
  kBase<<<(MROWS + 255) / 256, 256, 0, stream>>>(smp_out, Lq1, Lq0, lpz1, lpz0, blq, blp);

  // G_rest (raw acc) and ps/xouts (sigmoid/sample epilogue); 2560 blocks each
  kGemmBig<0><<<NT_M * 8, 256, 0, stream>>>(szr, V_h, x_bias, grest, nullptr, nullptr);
  kGemmBig<1><<<NT_M * 8, 256, 0, stream>>>(szf, V_h, x_bias, nullptr, ps_out, xo_out);

  for (int t = 0; t < TT; ++t) {
    kRowReduce<<<1, 256, 0, stream>>>(Cmat, t);
    kProj<<<(MROWS * 32 + 255) / 256, 256, 0, stream>>>(smp_out, Cmat, proj_h);
    kPerT<<<(MROWS + 255) / 256, 256, 0, stream>>>(proj_h, Lq1, Lq0, lpz1, lpz0, blq, blp, lqt, lpzt);
    kRankUpdate<<<(NT_M * NT_N / 2) / 8, 256, 0, stream>>>(proj_h, V_h, x_bias, xin, grest, part);
    kSumPartial<<<(MROWS + 255) / 256, 256, 0, stream>>>(part, lpx);
    kTerm<<<1, BB, 0, stream>>>(lqt, lpzt, lpx, pelbo);
  }
  kFinal<<<1, 1, 0, stream>>>(pelbo, loss);
}
